// RAFT_21131239096405
// MI455X (gfx1250) — matrix operand order repaired and checked by reading, not compiled, not run on hardware
//
#include <hip/hip_runtime.h>

// ---------------------------------------------------------------------------
// Types for CDNA5 WMMA (wave32): bf16 A/B fragments (16 x bf16), f32 C/D (8 x f32)
// ---------------------------------------------------------------------------
typedef __attribute__((ext_vector_type(16))) __bf16 v16bf;
typedef __attribute__((ext_vector_type(8)))  float  v8f;

__device__ __forceinline__ float act_apply(float v, int act) {
  if (act == 1) return v > 0.0f ? v : 0.0f;                 // relu
  if (act == 2) return 1.0f / (1.0f + __expf(-v));          // sigmoid
  if (act == 3) return tanhf(v);                            // tanh
  return v;
}

// ---------------------------------------------------------------------------
// Fragment-order layouts (global memory, no LDS needed in the GEMM):
//   A: [mt][kb][lane][16]  lane=(m&15)+16*kh, elems e: k = kb*32 + kh*16 + e
//   B: [nt][kb][lane][16]  lane=(n&15)+16*kh, elems e: k = kb*32 + kh*16 + e
// Each lane fragment = 32 contiguous bytes -> two global_load_b128 per fragment.
// K zero-padded to mult of 32 (KB blocks); N zero-padded to mult of 128 tiles.
// M (batched over images) is always a multiple of 64 in this network.
// ---------------------------------------------------------------------------

// GEMM: C[M,N] = act((A*B + bias) * scale).  256 thr = 8 waves,
// block tile 64x128, each wave a 2x2 grid of 16x16 WMMA tiles.
__global__ __launch_bounds__(256) void k_gemm_frag(
    const __bf16* __restrict__ Afrag, const __bf16* __restrict__ Bfrag,
    float* __restrict__ C, const float* __restrict__ bias,
    int M, int N, int KB, float scale, int act)
{
  const int tid  = threadIdx.x;
  const int wid  = tid >> 5;
  const int lane = tid & 31;
  const int m0   = lane & 15;
  const int kh   = lane >> 4;
  const int mt0  = blockIdx.y * 4 + (wid >> 2) * 2;   // 16-row tile index
  const int nt0  = blockIdx.x * 8 + (wid & 3) * 2;    // 16-col tile index

  const v16bf* A0 = (const v16bf*)Afrag + ((size_t)(mt0 + 0) * KB) * 32 + lane;
  const v16bf* A1 = (const v16bf*)Afrag + ((size_t)(mt0 + 1) * KB) * 32 + lane;
  const v16bf* B0 = (const v16bf*)Bfrag + ((size_t)(nt0 + 0) * KB) * 32 + lane;
  const v16bf* B1 = (const v16bf*)Bfrag + ((size_t)(nt0 + 1) * KB) * 32 + lane;

  v8f acc[2][2] = {};

  #pragma unroll 2
  for (int kb = 0; kb < KB; ++kb) {
    v16bf a0 = A0[(size_t)kb * 32];
    v16bf a1 = A1[(size_t)kb * 32];
    v16bf b0 = B0[(size_t)kb * 32];
    v16bf b1 = B1[(size_t)kb * 32];
    if (kb + 2 < KB) {
      __builtin_prefetch((const void*)&A0[(size_t)(kb + 2) * 32], 0, 1);
      __builtin_prefetch((const void*)&B0[(size_t)(kb + 2) * 32], 0, 1);
    }
    acc[0][0] = __builtin_amdgcn_wmma_f32_16x16x32_bf16(false, a0, false, b0, (short)0, acc[0][0], false, false);
    acc[0][1] = __builtin_amdgcn_wmma_f32_16x16x32_bf16(false, a0, false, b1, (short)0, acc[0][1], false, false);
    acc[1][0] = __builtin_amdgcn_wmma_f32_16x16x32_bf16(false, a1, false, b0, (short)0, acc[1][0], false, false);
    acc[1][1] = __builtin_amdgcn_wmma_f32_16x16x32_bf16(false, a1, false, b1, (short)0, acc[1][1], false, false);
  }

  // Epilogue: C/D layout -> row = tile*16 + kh*8 + i, col = tile*16 + m0
  #pragma unroll
  for (int im = 0; im < 2; ++im) {
    #pragma unroll
    for (int in2 = 0; in2 < 2; ++in2) {
      #pragma unroll
      for (int i = 0; i < 8; ++i) {
        int row = (mt0 + im) * 16 + kh * 8 + i;
        int col = (nt0 + in2) * 16 + m0;
        if (row < M && col < N) {
          float v = acc[im][in2][i];
          if (bias) v += bias[col];
          v *= scale;
          C[(size_t)row * N + col] = act_apply(v, act);
        }
      }
    }
  }
}

// Batched im2col (source [B,H,W,C] f32) directly into A-fragment order.
// Mper = Ho*Wo per image; global row m -> image b = m/Mper.
__global__ void k_im2col_frag(const float* __restrict__ src, __bf16* __restrict__ dst,
                              int H, int W, int C, int KH, int KW, int stride,
                              int padH, int padW, int Wo, int Mper, int KB, long long total)
{
  long long idx = (long long)blockIdx.x * blockDim.x + threadIdx.x;
  if (idx >= total) return;
  int e = (int)(idx & 15);
  long long t = idx >> 4;
  int lane = (int)(t & 31); t >>= 5;
  int kb = (int)(t % KB);
  int mt = (int)(t / KB);
  int kh = lane >> 4;
  int m  = mt * 16 + (lane & 15);
  int kk = kh * 16 + e;
  int k  = kb * 32 + kk;
  int K  = KH * KW * C;
  float v = 0.0f;
  if (k < K) {
    int b  = m / Mper;
    int ml = m % Mper;
    int c = k % C;
    int kyx = k / C;
    int kx = kyx % KW, ky = kyx / KW;
    int x = ml % Wo, y = ml / Wo;
    int sy = y * stride - padH + ky;
    int sx = x * stride - padW + kx;
    if (sy >= 0 && sy < H && sx >= 0 && sx < W)
      v = src[(size_t)b * H * W * C + ((size_t)sy * W + sx) * C + c];
  }
  dst[idx] = (__bf16)v;
}

// OIHW f32 weights -> B-fragment order, zero-padded K and N.
__global__ void k_pack_w_frag(const float* __restrict__ w, __bf16* __restrict__ dst,
                              int O, int C, int KH, int KW, int KB, long long total)
{
  long long idx = (long long)blockIdx.x * blockDim.x + threadIdx.x;
  if (idx >= total) return;
  int e = (int)(idx & 15);
  long long t = idx >> 4;
  int lane = (int)(t & 31); t >>= 5;
  int kb = (int)(t % KB);
  int nt = (int)(t / KB);
  int k = kb * 32 + (lane >> 4) * 16 + e;
  int n = nt * 16 + (lane & 15);
  int K = KH * KW * C;
  float v = 0.0f;
  if (k < K && n < O) {
    int c = k % C;
    int kyx = k / C;
    int kx = kyx % KW, ky = kyx / KW;
    v = w[(((size_t)n * C + c) * KH + ky) * KW + kx];
  }
  dst[idx] = (__bf16)v;
}

// [R,K] f32 row-major -> B-fragment order of B[k][n]=src[n*K+k] (correlation f2^T)
__global__ void k_pack_bt_frag(const float* __restrict__ src, __bf16* __restrict__ dst,
                               int R, int K, int KB, long long total)
{
  long long idx = (long long)blockIdx.x * blockDim.x + threadIdx.x;
  if (idx >= total) return;
  int e = (int)(idx & 15);
  long long t = idx >> 4;
  int lane = (int)(t & 31); t >>= 5;
  int kb = (int)(t % KB);
  int nt = (int)(t / KB);
  int k = kb * 32 + (lane >> 4) * 16 + e;
  int n = nt * 16 + (lane & 15);
  float v = (k < K && n < R) ? src[(size_t)n * K + k] : 0.0f;
  dst[idx] = (__bf16)v;
}

// ---------------------------------------------------------------------------
// Batched instance norm (+relu) over HW, HWC layout; block = (image, channel)
// ---------------------------------------------------------------------------
__global__ __launch_bounds__(256) void k_inorm_relu(float* __restrict__ xall, int HW, int C)
{
  __shared__ float ss[256], ss2[256];
  int c = blockIdx.x % C;
  float* x = xall + (size_t)(blockIdx.x / C) * HW * C;
  int tid = threadIdx.x;
  float s = 0.f, s2 = 0.f;
  for (int i = tid; i < HW; i += 256) {
    float v = x[(size_t)i * C + c];
    s += v; s2 += v * v;
  }
  ss[tid] = s; ss2[tid] = s2;
  __syncthreads();
  for (int st = 128; st > 0; st >>= 1) {
    if (tid < st) { ss[tid] += ss[tid + st]; ss2[tid] += ss2[tid + st]; }
    __syncthreads();
  }
  float mean = ss[0] / (float)HW;
  float var  = ss2[0] / (float)HW - mean * mean;
  float inv  = rsqrtf(var + 1e-5f);
  for (int i = tid; i < HW; i += 256) {
    size_t id = (size_t)i * C + c;
    float v = (x[id] - mean) * inv;
    x[id] = v > 0.f ? v : 0.f;
  }
}

// ---------------------------------------------------------------------------
// Pointwise / misc kernels (all batched over images where relevant)
// ---------------------------------------------------------------------------
__global__ void k_normalize(const float* __restrict__ src, float* __restrict__ dst, int n)
{
  int i = blockIdx.x * blockDim.x + threadIdx.x;
  if (i < n) dst[i] = 2.0f * (src[i] / 255.0f) - 1.0f;
}

__global__ void k_split_net_inp(const float* __restrict__ cnet,
                                float* __restrict__ net, float* __restrict__ inp, int P)
{
  int idx = blockIdx.x * blockDim.x + threadIdx.x;
  if (idx >= P * 128) return;
  int p = idx / 128, c = idx % 128;
  net[idx] = tanhf(cnet[(size_t)p * 256 + c]);
  float v = cnet[(size_t)p * 256 + 128 + c];
  inp[idx] = v > 0.f ? v : 0.f;
}

__global__ void k_concat2(const float* __restrict__ a, int Ca,
                          const float* __restrict__ b, int Cb,
                          float* __restrict__ out, int P)
{
  int Ct = Ca + Cb;
  long long total = (long long)P * Ct;
  long long idx = (long long)blockIdx.x * blockDim.x + threadIdx.x;
  if (idx >= total) return;
  int p = (int)(idx / Ct), c = (int)(idx % Ct);
  out[idx] = (c < Ca) ? a[(size_t)p * Ca + c] : b[(size_t)p * Cb + (c - Ca)];
}

// hq = concat(r*net, x) with r,net 128ch and x 256ch
__global__ void k_rnet(const float* __restrict__ r, const float* __restrict__ net,
                       const float* __restrict__ x, float* __restrict__ hq, int P)
{
  long long total = (long long)P * 384;
  long long idx = (long long)blockIdx.x * blockDim.x + threadIdx.x;
  if (idx >= total) return;
  int p = (int)(idx / 384), c = (int)(idx % 384);
  hq[idx] = (c < 128) ? r[(size_t)p * 128 + c] * net[(size_t)p * 128 + c]
                      : x[(size_t)p * 256 + (c - 128)];
}

__global__ void k_gru_update(float* __restrict__ net, const float* __restrict__ z,
                             const float* __restrict__ q, int n)
{
  int i = blockIdx.x * blockDim.x + threadIdx.x;
  if (i < n) net[i] = (1.0f - z[i]) * net[i] + z[i] * q[i];
}

__global__ void k_coords_init(float* __restrict__ coords, int total, int Mper, int W)
{
  int p = blockIdx.x * blockDim.x + threadIdx.x;
  if (p >= total) return;
  int pl = p % Mper;
  coords[p * 2 + 0] = (float)(pl % W);
  coords[p * 2 + 1] = (float)(pl / W);
}

__global__ void k_flow(const float* __restrict__ coords, float* __restrict__ flow,
                       int total, int Mper, int W)
{
  int p = blockIdx.x * blockDim.x + threadIdx.x;
  if (p >= total) return;
  int pl = p % Mper;
  flow[p * 2 + 0] = coords[p * 2 + 0] - (float)(pl % W);
  flow[p * 2 + 1] = coords[p * 2 + 1] - (float)(pl / W);
}

__global__ void k_addflow(float* __restrict__ coords, const float* __restrict__ dflow, int n)
{
  int i = blockIdx.x * blockDim.x + threadIdx.x;
  if (i < n) coords[i] += dflow[i];
}

__global__ void k_avgpool2(const float* __restrict__ in, float* __restrict__ out,
                           int R, int H, int W)
{
  int Ho = H >> 1, Wo = W >> 1;
  long long total = (long long)R * Ho * Wo;
  long long idx = (long long)blockIdx.x * blockDim.x + threadIdx.x;
  if (idx >= total) return;
  int x = (int)(idx % Wo);
  long long t = idx / Wo;
  int y = (int)(t % Ho);
  int r = (int)(t / Ho);
  const float* b = in + (size_t)r * H * W;
  out[idx] = 0.25f * (b[(2 * y) * W + 2 * x] + b[(2 * y) * W + 2 * x + 1] +
                      b[(2 * y + 1) * W + 2 * x] + b[(2 * y + 1) * W + 2 * x + 1]);
}

// ---------------------------------------------------------------------------
// Correlation pyramid lookup (batched): pyramids are [B*1600, Hh*Ww] row-major,
// global pixel p indexes rows directly. Bilinear, zero padding, 4 lvls x 81 off.
// ---------------------------------------------------------------------------
__global__ void k_corr_lookup(const float* __restrict__ p0, const float* __restrict__ p1,
                              const float* __restrict__ p2, const float* __restrict__ p3,
                              const float* __restrict__ coords, float* __restrict__ feat,
                              int total, int H8, int W8)
{
  long long tot = (long long)total * 4 * 81;
  long long idx = (long long)blockIdx.x * blockDim.x + threadIdx.x;
  if (idx >= tot) return;
  int j = (int)(idx % 81);
  long long t = idx / 81;
  int lvl = (int)(t & 3);
  int p = (int)(t >> 2);

  const float* img = (lvl == 0) ? p0 : (lvl == 1) ? p1 : (lvl == 2) ? p2 : p3;
  int Hh = H8 >> lvl, Ww = W8 >> lvl;
  float inv = 1.0f / (float)(1 << lvl);
  float cx = coords[p * 2 + 0] * inv + (float)(j % 9 - 4);
  float cy = coords[p * 2 + 1] * inv + (float)(j / 9 - 4);

  float x0 = floorf(cx), y0 = floorf(cy);
  float wx = cx - x0, wy = cy - y0;
  const float* base = img + (size_t)p * Hh * Ww;

  auto samp = [&](float xi, float yi) -> float {
    if (xi < 0.f || xi > (float)(Ww - 1) || yi < 0.f || yi > (float)(Hh - 1)) return 0.f;
    return base[(int)yi * Ww + (int)xi];
  };

  float v = (1.f - wx) * (1.f - wy) * samp(x0, y0)
          + (1.f - wx) * wy         * samp(x0, y0 + 1.f)
          + wx         * (1.f - wy) * samp(x0 + 1.f, y0)
          + wx         * wy         * samp(x0 + 1.f, y0 + 1.f);
  feat[(size_t)p * 324 + lvl * 81 + j] = v;
}

// ---------------------------------------------------------------------------
// Convex upsample (batched): mask [P,576=(9,8,8)], flow = 8*(coords1-grid)
// Writes d_out slices for all images at iteration `it`.
// ---------------------------------------------------------------------------
__global__ void k_upsample(const float* __restrict__ coords, const float* __restrict__ mask,
                           float* __restrict__ out, int it, int Mper, int H8, int W8, int total)
{
  int tid = blockIdx.x * blockDim.x + threadIdx.x;
  if (tid >= total * 64) return;
  int ij = tid & 63;
  int p = tid >> 6;
  int i = ij >> 3, j = ij & 7;
  int n = p / Mper;
  int pl = p % Mper;
  int y = pl / W8, x = pl % W8;

  float wv[9];
  float mx = -1e30f;
  #pragma unroll
  for (int k = 0; k < 9; ++k) {
    wv[k] = mask[(size_t)p * 576 + (k * 8 + i) * 8 + j];
    mx = fmaxf(mx, wv[k]);
  }
  float ssum = 0.f;
  #pragma unroll
  for (int k = 0; k < 9; ++k) { wv[k] = __expf(wv[k] - mx); ssum += wv[k]; }

  #pragma unroll
  for (int c = 0; c < 2; ++c) {
    float acc = 0.f;
    #pragma unroll
    for (int k = 0; k < 9; ++k) {
      int yy = y + k / 3 - 1;
      int xx = x + k % 3 - 1;
      if (yy >= 0 && yy < H8 && xx >= 0 && xx < W8) {
        int q = n * Mper + yy * W8 + xx;
        float f = 8.0f * (coords[q * 2 + c] - (float)(c == 0 ? xx : yy));
        acc += wv[k] * f;
      }
    }
    acc /= ssum;
    size_t o = ((((size_t)it * 6 + n) * 2 + c) * 320 + (size_t)(y * 8 + i)) * 320
             + (size_t)(x * 8 + j);
    out[o] = acc;
  }
}

// ---------------------------------------------------------------------------
// Host orchestration
// ---------------------------------------------------------------------------
struct LayerDef { int pidx; int O, C, KH, KW; };

enum { Lf1, Lf2, Lf3, Lf4, Lc1, Lc2, Lc3, Lc4,
       Lmc1, Lmc2, Lmf1, Lmf2, Lm,
       Lgz_h, Lgz_v, Lgr_h, Lgr_v, Lgq_h, Lgq_v,
       Lfh1, Lfh2, Lmk1, Lmk2, NLAYERS };

static const LayerDef LDEFS[NLAYERS] = {
  {2,  64,  1,   7, 7}, {4,  96,  64,  3, 3}, {6,  128, 96,  3, 3}, {8,  256, 128, 1, 1},
  {10, 64,  1,   7, 7}, {12, 96,  64,  3, 3}, {14, 128, 96,  3, 3}, {16, 256, 128, 1, 1},
  {18, 256, 324, 1, 1}, {20, 192, 256, 3, 3}, {22, 128, 2,   7, 7}, {24, 64,  128, 3, 3},
  {26, 126, 256, 3, 3},
  {28, 128, 384, 1, 5}, {30, 128, 384, 5, 1},
  {32, 128, 384, 1, 5}, {34, 128, 384, 5, 1},
  {36, 128, 384, 1, 5}, {38, 128, 384, 5, 1},
  {40, 256, 128, 3, 3}, {42, 2,   256, 3, 3},
  {44, 256, 128, 3, 3}, {46, 576, 256, 1, 1},
};

static inline int kpad32(int k) { return (k + 31) & ~31; }

extern "C" void kernel_launch(void* const* d_in, const int* in_sizes, int n_in,
                              void* d_out, int out_size, void* d_ws, size_t ws_size,
                              hipStream_t stream)
{
  (void)in_sizes; (void)n_in; (void)out_size; (void)ws_size;

  const int NB_IMG = 6, HWimg = 320 * 320;
  const int H8 = 40, W8 = 40, HW8 = 1600;
  const int P = NB_IMG * HW8;          // 9600 batched pixels at 1/8 res
  const int ITERS = 12;

  const float* image = (const float*)d_in[0];
  const float* templ = (const float*)d_in[1];
  float* out = (float*)d_out;

  // ---- bump allocator over workspace ----
  size_t off = 0;
  auto wsalloc = [&](size_t bytes) -> char* {
    char* p = (char*)d_ws + off;
    off += (bytes + 255) & ~(size_t)255;
    return p;
  };

  // ---- weight packs into B-fragment order (once per launch, deterministic) ----
  __bf16* wpf[NLAYERS];
  for (int li = 0; li < NLAYERS; ++li) {
    const LayerDef& L = LDEFS[li];
    int K = L.KH * L.KW * L.C;
    int KB = kpad32(K) / 32;
    int NBb = (L.O + 127) / 128;        // 128-col GEMM blocks
    long long total = (long long)(NBb * 8) * KB * 512;
    wpf[li] = (__bf16*)wsalloc((size_t)total * sizeof(__bf16));
    k_pack_w_frag<<<dim3((unsigned)((total + 255) / 256)), 256, 0, stream>>>(
        (const float*)d_in[L.pidx], wpf[li], L.O, L.C, L.KH, L.KW, KB, total);
  }

  // ---- batched persistent buffers ----
  float* pyr0   = (float*)wsalloc((size_t)P * 1600 * 4);
  float* pyr1   = (float*)wsalloc((size_t)P * 400 * 4);
  float* pyr2   = (float*)wsalloc((size_t)P * 100 * 4);
  float* pyr3   = (float*)wsalloc((size_t)P * 25 * 4);
  float* netb   = (float*)wsalloc((size_t)P * 128 * 4);
  float* inpb   = (float*)wsalloc((size_t)P * 128 * 4);
  float* coordsb= (float*)wsalloc((size_t)P * 2 * 4);

  // ---- shared scratch (batched) ----
  const size_t COLMAX = 22118400; // max M*Kpad elems: 38400x576 / 9600x2304
  float*  imgAll = (float*)wsalloc((size_t)NB_IMG * HWimg * 4);
  float*  a1    = (float*)wsalloc((size_t)NB_IMG * 25600 * 64 * 4);
  float*  a2    = (float*)wsalloc((size_t)NB_IMG * 6400 * 96 * 4);
  float*  a3    = (float*)wsalloc((size_t)P * 128 * 4);
  float*  feaA  = (float*)wsalloc((size_t)P * 256 * 4);
  float*  feaB  = (float*)wsalloc((size_t)P * 256 * 4);
  float*  cnetb = (float*)wsalloc((size_t)P * 256 * 4);
  __bf16* colbuf = (__bf16*)wsalloc(COLMAX * sizeof(__bf16));
  __bf16* f1b   = (__bf16*)wsalloc((size_t)P * 256 * sizeof(__bf16));
  __bf16* f2t   = (__bf16*)wsalloc((size_t)104 * 8 * 512 * sizeof(__bf16)); // NTpad=104, KB=8
  float*  feat  = (float*)wsalloc((size_t)P * 324 * 4);
  float*  c1    = (float*)wsalloc((size_t)P * 256 * 4);
  float*  c2    = (float*)wsalloc((size_t)P * 192 * 4);
  float*  fm1   = (float*)wsalloc((size_t)P * 128 * 4);
  float*  fm2   = (float*)wsalloc((size_t)P * 64 * 4);
  float*  cf    = (float*)wsalloc((size_t)P * 256 * 4);
  float*  mo    = (float*)wsalloc((size_t)P * 126 * 4);
  float*  motion= (float*)wsalloc((size_t)P * 128 * 4);
  float*  xb    = (float*)wsalloc((size_t)P * 256 * 4);
  float*  hx    = (float*)wsalloc((size_t)P * 384 * 4);
  float*  zb    = (float*)wsalloc((size_t)P * 128 * 4);
  float*  rb    = (float*)wsalloc((size_t)P * 128 * 4);
  float*  qb    = (float*)wsalloc((size_t)P * 128 * 4);
  float*  hq    = (float*)wsalloc((size_t)P * 384 * 4);
  float*  t1    = (float*)wsalloc((size_t)P * 256 * 4);
  float*  dflow = (float*)wsalloc((size_t)P * 2 * 4);
  float*  t2    = (float*)wsalloc((size_t)P * 256 * 4);
  float*  maskb = (float*)wsalloc((size_t)P * 576 * 4);
  float*  flow  = (float*)wsalloc((size_t)P * 2 * 4);

  // ---- conv = batched fragment-order im2col + WMMA GEMM ----
  auto conv2d = [&](const float* in, int H, int W, int C, int li,
                    int stride, int padH, int padW, int act, float scale, float* outp) {
    const LayerDef& L = LDEFS[li];
    int Ho = (H + 2 * padH - L.KH) / stride + 1;
    int Wo = (W + 2 * padW - L.KW) / stride + 1;
    int K = L.KH * L.KW * C;
    int KB = kpad32(K) / 32;
    int Mper = Ho * Wo;
    int M = NB_IMG * Mper;                       // batched rows, multiple of 64
    long long totA = (long long)(M / 16) * KB * 512;
    k_im2col_frag<<<dim3((unsigned)((totA + 255) / 256)), 256, 0, stream>>>(
        in, colbuf, H, W, C, L.KH, L.KW, stride, padH, padW, Wo, Mper, KB, totA);
    dim3 g((L.O + 127) / 128, M / 64);
    k_gemm_frag<<<g, 256, 0, stream>>>(colbuf, wpf[li], outp,
                                       (const float*)d_in[L.pidx + 1],
                                       M, L.O, KB, scale, act);
  };

  auto encoder = [&](const float* in_img, int baseL, bool use_in, float* outp) {
    conv2d(in_img, 320, 320, 1, baseL + 0, 2, 3, 3, use_in ? 0 : 1, 1.f, a1);
    if (use_in) k_inorm_relu<<<NB_IMG * 64, 256, 0, stream>>>(a1, 160 * 160, 64);
    conv2d(a1, 160, 160, 64, baseL + 1, 2, 1, 1, use_in ? 0 : 1, 1.f, a2);
    if (use_in) k_inorm_relu<<<NB_IMG * 96, 256, 0, stream>>>(a2, 80 * 80, 96);
    conv2d(a2, 80, 80, 96, baseL + 2, 2, 1, 1, use_in ? 0 : 1, 1.f, a3);
    if (use_in) k_inorm_relu<<<NB_IMG * 128, 256, 0, stream>>>(a3, 40 * 40, 128);
    conv2d(a3, 40, 40, 128, baseL + 3, 1, 0, 0, 0, 1.f, outp);
  };

  // ---- encoders (whole 6-image batch at once) ----
  k_normalize<<<dim3((NB_IMG * HWimg + 255) / 256), 256, 0, stream>>>(image, imgAll, NB_IMG * HWimg);
  encoder(imgAll, Lf1, true, feaA);        // fnet(images)
  encoder(imgAll, Lc1, false, cnetb);      // cnet(images)
  k_split_net_inp<<<dim3((P * 128 + 255) / 256), 256, 0, stream>>>(cnetb, netb, inpb, P);

  k_normalize<<<dim3((NB_IMG * HWimg + 255) / 256), 256, 0, stream>>>(templ, imgAll, NB_IMG * HWimg);
  encoder(imgAll, Lf1, true, feaB);        // fnet(templates)

  // ---- correlation pyramid: per-image GEMM into batched row-sliced volume ----
  {
    const int KBc = 8;                   // K = 256
    long long totA = (long long)(P / 16) * KBc * 512;   // batched f1 fragments
    k_im2col_frag<<<dim3((unsigned)((totA + 255) / 256)), 256, 0, stream>>>(
        feaA, f1b, 40, 40, 256, 1, 1, 1, 0, 0, 40, HW8, KBc, totA);
    long long totB = (long long)104 * KBc * 512;        // NTpad = ceil(1600/128)*8
    for (int n = 0; n < NB_IMG; ++n) {
      k_pack_bt_frag<<<dim3((unsigned)((totB + 255) / 256)), 256, 0, stream>>>(
          feaB + (size_t)n * HW8 * 256, f2t, HW8, 256, KBc, totB);
      dim3 g((HW8 + 127) / 128, HW8 / 64);
      k_gemm_frag<<<g, 256, 0, stream>>>(
          f1b + (size_t)n * 100 * KBc * 512,            // image-n row tiles
          f2t, pyr0 + (size_t)n * HW8 * 1600,
          (const float*)nullptr, HW8, HW8, KBc, 0.0625f, 0);
    }
    k_avgpool2<<<dim3((unsigned)(((long long)P * 400 + 255) / 256)), 256, 0, stream>>>(pyr0, pyr1, P, 40, 40);
    k_avgpool2<<<dim3((unsigned)(((long long)P * 100 + 255) / 256)), 256, 0, stream>>>(pyr1, pyr2, P, 20, 20);
    k_avgpool2<<<dim3((unsigned)(((long long)P * 25 + 255) / 256)), 256, 0, stream>>>(pyr2, pyr3, P, 10, 10);
  }

  k_coords_init<<<dim3((P + 255) / 256), 256, 0, stream>>>(coordsb, P, HW8, W8);

  // ---- GRU direction pass (batched) ----
  auto gru_dir = [&](int lz, int lr, int lq, int padH, int padW) {
    k_concat2<<<dim3((unsigned)(((long long)P * 384 + 255) / 256)), 256, 0, stream>>>(netb, 128, xb, 256, hx, P);
    conv2d(hx, 40, 40, 384, lz, 1, padH, padW, 2, 1.f, zb);
    conv2d(hx, 40, 40, 384, lr, 1, padH, padW, 2, 1.f, rb);
    k_rnet<<<dim3((unsigned)(((long long)P * 384 + 255) / 256)), 256, 0, stream>>>(rb, netb, xb, hq, P);
    conv2d(hq, 40, 40, 384, lq, 1, padH, padW, 3, 1.f, qb);
    k_gru_update<<<dim3((P * 128 + 255) / 256), 256, 0, stream>>>(netb, zb, qb, P * 128);
  };

  // ---- iterative refinement (all 6 images batched per iteration) ----
  for (int it = 0; it < ITERS; ++it) {
    k_corr_lookup<<<dim3((unsigned)(((long long)P * 324 + 255) / 256)), 256, 0, stream>>>(
        pyr0, pyr1, pyr2, pyr3, coordsb, feat, P, H8, W8);
    k_flow<<<dim3((P + 255) / 256), 256, 0, stream>>>(coordsb, flow, P, HW8, W8);

    // motion encoder
    conv2d(feat, 40, 40, 324, Lmc1, 1, 0, 0, 1, 1.f, c1);
    conv2d(c1,   40, 40, 256, Lmc2, 1, 1, 1, 1, 1.f, c2);
    conv2d(flow, 40, 40, 2,   Lmf1, 1, 3, 3, 1, 1.f, fm1);
    conv2d(fm1,  40, 40, 128, Lmf2, 1, 1, 1, 1, 1.f, fm2);
    k_concat2<<<dim3((unsigned)(((long long)P * 256 + 255) / 256)), 256, 0, stream>>>(c2, 192, fm2, 64, cf, P);
    conv2d(cf,   40, 40, 256, Lm,   1, 1, 1, 1, 1.f, mo);
    k_concat2<<<dim3((unsigned)(((long long)P * 128 + 255) / 256)), 256, 0, stream>>>(mo, 126, flow, 2, motion, P);
    k_concat2<<<dim3((unsigned)(((long long)P * 256 + 255) / 256)), 256, 0, stream>>>(inpb, 128, motion, 128, xb, P);

    // SepConvGRU: horizontal (1x5, pad (0,2)) then vertical (5x1, pad (2,0))
    gru_dir(Lgz_h, Lgr_h, Lgq_h, 0, 2);
    gru_dir(Lgz_v, Lgr_v, Lgq_v, 2, 0);

    // flow head + mask head
    conv2d(netb, 40, 40, 128, Lfh1, 1, 1, 1, 1, 1.f,   t1);
    conv2d(t1,   40, 40, 256, Lfh2, 1, 1, 1, 0, 1.f,   dflow);
    conv2d(netb, 40, 40, 128, Lmk1, 1, 1, 1, 1, 1.f,   t2);
    conv2d(t2,   40, 40, 256, Lmk2, 1, 0, 0, 0, 0.25f, maskb);

    k_addflow<<<dim3((P * 2 + 255) / 256), 256, 0, stream>>>(coordsb, dflow, P * 2);
    k_upsample<<<dim3((P * 64 + 255) / 256), 256, 0, stream>>>(
        coordsb, maskb, out, it, HW8, H8, W8, P);
  }
}
